// NOCD_DL_75763223102021
// MI455X (gfx1250) — matrix operand order repaired and checked by reading, not compiled, not run on hardware
//
#include <hip/hip_runtime.h>
#include <hip/hip_bf16.h>
#include <math.h>

// ---------------------------------------------------------------------------
// Problem constants (from reference)
// ---------------------------------------------------------------------------
#define NNODES 20000
#define DIN    1000
#define NEDGES 640000

static inline int pad32(int x) { return (x + 31) & ~31; }
static inline int pad64(int x) { return (x + 63) & ~63; }

// ---------------------------------------------------------------------------
// WMMA types (CDNA5 gfx1250, wave32)
// ---------------------------------------------------------------------------
typedef __attribute__((ext_vector_type(16))) __bf16 v16bf;
typedef __attribute__((ext_vector_type(8)))  __bf16 v8bf;
typedef __attribute__((ext_vector_type(8)))  float  v8f;

__device__ __forceinline__ __bf16 f2bf(float f) {
  unsigned u = __builtin_bit_cast(unsigned, f);
  unsigned r = u + 0x7FFFu + ((u >> 16) & 1u);   // round-to-nearest-even
  unsigned short h = (unsigned short)(r >> 16);
  return __builtin_bit_cast(__bf16, h);
}
__device__ __forceinline__ __bf16 bf_zero() {
  return __builtin_bit_cast(__bf16, (unsigned short)0);
}

// ---------------------------------------------------------------------------
// WMMA GEMM with async-to-LDS A staging (double buffered):
//   C[M,Nc](f32,ldc) = A[M,Kpad](bf16,lda=Kpad) @ B (stored transposed as
//   BT[Ncpad64,Kpad](bf16,ldb=Kpad)), + bias, optional relu.
//
// Invariants established by the staging kernels (no bounds checks needed):
//   - K padded to multiple of 32 with zeros (A columns and BT columns)
//   - Nc padded to multiple of 64 with zero rows in BT
//   - M % 16 == 0 (M = 20000), so each wave's 16-row tile is all-in or all-out
//
// Block: 256 thr = 8 waves -> 64(M) x 64(N); wave = 16x32 via 2 accumulators.
// A tile (64 rows x 32 K bf16 = 4 KB) is staged into LDS with
// GLOBAL_LOAD_ASYNC_TO_LDS_B128 (one b128 per thread), synchronized with
// s_wait_asynccnt + workgroup barrier, and consumed via ds_load_b128.
// Wave w stages tile rows 8w..8w+7; its output tile rows are (w>>1)*16..+15,
// so a wave's staged rows are in-bounds iff its output rows are -> the
// early-exit for the ragged last row-block stays consistent.
//
// Fragment layouts per CDNA5 ISA 7.12.2:
//   A 16x32: lane r=lane&15, h=lane>>4; elems 0..7 -> K=k0+h*8+e,
//            elems 8..15 -> K=k0+16+h*8+(e-8)
//   B 32x16: lane c=lane&15, h=lane>>4; same K mapping as A (2x b128 via BT)
//   C 16x16: elem v -> row = v + 8*h, col = lane&15
// ---------------------------------------------------------------------------
__global__ __launch_bounds__(256) void k_gemm_bf16_wmma(
    const __bf16* __restrict__ A, int lda,
    const __bf16* __restrict__ BT, int ldb,
    const float* __restrict__ bias,
    float* __restrict__ C, int ldc,
    int M, int Kpad, int Nc, int do_relu)
{
  __shared__ __align__(16) __bf16 As[2][64 * 32];   // 8 KB double buffer

  const int lane = threadIdx.x & 31;
  const int wave = threadIdx.x >> 5;
  const int wm = wave >> 1;                  // 0..3
  const int wn = wave & 1;                   // 0..1
  const int rowblk = blockIdx.y * 64;
  const int row0 = rowblk + wm * 16;
  const int col0 = blockIdx.x * 64 + wn * 32;
  if (row0 + 16 > M) return;                 // wave-uniform (M % 16 == 0)

  const int l15 = lane & 15;
  const int h   = lane >> 4;

  // staging assignment: thread t copies 16B = row t/4, K-offset (t%4)*8
  const int trow = threadIdx.x >> 2;
  const int tcol = (threadIdx.x & 3) * 8;
  const __bf16* gsrc = A + (size_t)(rowblk + trow) * lda + tcol;
  const unsigned lds0 = (unsigned)(uintptr_t)&As[0][trow * 32 + tcol];
  const unsigned lds1 = (unsigned)(uintptr_t)&As[1][trow * 32 + tcol];

  const __bf16* Bt0 = BT + (size_t)(col0 + l15) * ldb;
  const __bf16* Bt1 = BT + (size_t)(col0 + 16 + l15) * ldb;

  // prologue: stage k0 = 0 into buffer 0 (async DMA, ASYNCcnt-tracked)
  asm volatile("global_load_async_to_lds_b128 %0, %1, off"
               :: "v"(lds0), "v"(gsrc) : "memory");

  v8f acc0 = {}, acc1 = {};
  int cur = 0;

  for (int k0 = 0; k0 < Kpad; k0 += 32) {
    // make this k-step's staged A tile visible to the whole workgroup
    asm volatile("s_wait_asynccnt 0x0" ::: "memory");
    __syncthreads();

    // kick off async staging of the next tile into the other buffer
    if (k0 + 32 < Kpad) {
      const __bf16* g = gsrc + (k0 + 32);
      unsigned l = cur ? lds0 : lds1;
      asm volatile("global_load_async_to_lds_b128 %0, %1, off"
                   :: "v"(l), "v"(g) : "memory");
    }

    // A fragment from LDS (ds_load_b128 x2)
    const __bf16* as = &As[cur][(wm * 16 + l15) * 32];
    v8bf aL = *reinterpret_cast<const v8bf*>(as + h * 8);
    v8bf aH = *reinterpret_cast<const v8bf*>(as + 16 + h * 8);
    v16bf a = __builtin_shufflevector(aL, aH,
        0, 1, 2, 3, 4, 5, 6, 7, 8, 9, 10, 11, 12, 13, 14, 15);

    // B fragments straight from global (weights are small / L2-resident)
    v8bf b0L = *reinterpret_cast<const v8bf*>(Bt0 + k0 + h * 8);
    v8bf b0H = *reinterpret_cast<const v8bf*>(Bt0 + k0 + 16 + h * 8);
    v16bf b0 = __builtin_shufflevector(b0L, b0H,
        0, 1, 2, 3, 4, 5, 6, 7, 8, 9, 10, 11, 12, 13, 14, 15);

    v8bf b1L = *reinterpret_cast<const v8bf*>(Bt1 + k0 + h * 8);
    v8bf b1H = *reinterpret_cast<const v8bf*>(Bt1 + k0 + 16 + h * 8);
    v16bf b1 = __builtin_shufflevector(b1L, b1H,
        0, 1, 2, 3, 4, 5, 6, 7, 8, 9, 10, 11, 12, 13, 14, 15);

    acc0 = __builtin_amdgcn_wmma_f32_16x16x32_bf16(
        false, a, false, b0, (short)0, acc0, false, false);
    acc1 = __builtin_amdgcn_wmma_f32_16x16x32_bf16(
        false, a, false, b1, (short)0, acc1, false, false);

    cur ^= 1;
  }

  // store: rows guaranteed in-bounds; columns checked against true Nc
  {
    int cc = col0 + l15;
    if (cc < Nc) {
      float bv = bias ? bias[cc] : 0.0f;
#pragma unroll
      for (int v = 0; v < 8; ++v) {
        int rr = row0 + v + 8 * h;
        float val = acc0[v] + bv;
        if (do_relu) val = fmaxf(val, 0.0f);
        C[(size_t)rr * ldc + cc] = val;
      }
    }
    cc = col0 + 16 + l15;
    if (cc < Nc) {
      float bv = bias ? bias[cc] : 0.0f;
#pragma unroll
      for (int v = 0; v < 8; ++v) {
        int rr = row0 + v + 8 * h;
        float val = acc1[v] + bv;
        if (do_relu) val = fmaxf(val, 0.0f);
        C[(size_t)rr * ldc + cc] = val;
      }
    }
  }
}

// ---------------------------------------------------------------------------
// f32 [rows,cols] (row stride lds) -> bf16 [rows,Kpad], zero-padded columns
// ---------------------------------------------------------------------------
__global__ void k_f2bf_pad(const float* __restrict__ src, int lds, int cols,
                           __bf16* __restrict__ dst, int Kpad, int rows)
{
  size_t i = (size_t)blockIdx.x * blockDim.x + threadIdx.x;
  size_t n = (size_t)rows * Kpad;
  if (i >= n) return;
  int r = (int)(i / Kpad), c = (int)(i % Kpad);
  dst[i] = (c < cols) ? f2bf(src[(size_t)r * lds + c]) : bf_zero();
}

// ---------------------------------------------------------------------------
// weight f32 [K,Nc] -> transposed padded bf16 [Ncpad, Kpad]
// ---------------------------------------------------------------------------
__global__ void k_w2bfT(const float* __restrict__ src, int K, int Nc,
                        __bf16* __restrict__ dst, int Kpad, int Ncpad)
{
  size_t i = (size_t)blockIdx.x * blockDim.x + threadIdx.x;
  size_t n = (size_t)Ncpad * Kpad;
  if (i >= n) return;
  int c = (int)(i / Kpad), k = (int)(i % Kpad);
  dst[i] = (c < Nc && k < K) ? f2bf(src[(size_t)k * Nc + c]) : bf_zero();
}

// ---------------------------------------------------------------------------
// SpMM scatter: agg[rows[e], :] += vals[e] * sup[cols[e], :]
// ---------------------------------------------------------------------------
__global__ void k_spmm(const int* __restrict__ er, const int* __restrict__ ec,
                       const float* __restrict__ ev,
                       const float* __restrict__ sup, float* __restrict__ agg,
                       int nc)
{
  int e = blockIdx.x;
  int c = blockIdx.y * blockDim.x + threadIdx.x;
  if (c >= nc) return;
  int rd = er[e], cs = ec[e];
  float v = ev[e];
  atomicAdd(&agg[(size_t)rd * nc + c], v * sup[(size_t)cs * nc + c]);
}

// elementwise elu(leaky_relu(x,0.2)) in place
__global__ void k_act(float* __restrict__ x, size_t n)
{
  size_t i = (size_t)blockIdx.x * blockDim.x + threadIdx.x;
  if (i >= n) return;
  float v = x[i];
  x[i] = (v > 0.0f) ? v : (expf(0.2f * v) - 1.0f);
}

// column mean / rstd (biased var), one block per column
__global__ __launch_bounds__(256) void k_colstats(
    const float* __restrict__ X, int ld, int rows,
    float* __restrict__ mean, float* __restrict__ rstd)
{
  int c = blockIdx.x;
  __shared__ float ss[256], sq[256];
  float s = 0.0f, q = 0.0f;
  for (int r = threadIdx.x; r < rows; r += 256) {
    float v = X[(size_t)r * ld + c];
    s += v; q += v * v;
  }
  ss[threadIdx.x] = s; sq[threadIdx.x] = q;
  __syncthreads();
  for (int st = 128; st > 0; st >>= 1) {
    if ((int)threadIdx.x < st) {
      ss[threadIdx.x] += ss[threadIdx.x + st];
      sq[threadIdx.x] += sq[threadIdx.x + st];
    }
    __syncthreads();
  }
  if (threadIdx.x == 0) {
    float m = ss[0] / rows;
    float var = sq[0] / rows - m * m;
    mean[c] = m;
    rstd[c] = rsqrtf(var + 1e-5f);
  }
}

__global__ void k_bnorm(const float* __restrict__ src, int lds,
                        float* __restrict__ dst, int ldd,
                        const float* __restrict__ mean,
                        const float* __restrict__ rstd,
                        int rows, int cols)
{
  size_t i = (size_t)blockIdx.x * blockDim.x + threadIdx.x;
  size_t n = (size_t)rows * cols;
  if (i >= n) return;
  int r = (int)(i / cols), c = (int)(i % cols);
  dst[(size_t)r * ldd + c] = (src[(size_t)r * lds + c] - mean[c]) * rstd[c];
}

__global__ void k_copy2d(const float* __restrict__ src, int lds,
                         float* __restrict__ dst, int ldd, int rows, int cols)
{
  size_t i = (size_t)blockIdx.x * blockDim.x + threadIdx.x;
  size_t n = (size_t)rows * cols;
  if (i >= n) return;
  int r = (int)(i / cols), c = (int)(i % cols);
  dst[(size_t)r * ldd + c] = src[(size_t)r * lds + c];
}

// tanh -> softmax(axis=1) -> L2 normalize, per row of U[rows,5], in place
__global__ void k_head(float* __restrict__ U, int rows)
{
  int r = blockIdx.x * blockDim.x + threadIdx.x;
  if (r >= rows) return;
  float v[5]; float mx = -1e30f;
#pragma unroll
  for (int j = 0; j < 5; ++j) { v[j] = tanhf(U[r * 5 + j]); mx = fmaxf(mx, v[j]); }
  float s = 0.0f;
#pragma unroll
  for (int j = 0; j < 5; ++j) { v[j] = expf(v[j] - mx); s += v[j]; }
  float inv = 1.0f / s, q = 0.0f;
#pragma unroll
  for (int j = 0; j < 5; ++j) { v[j] *= inv; q += v[j] * v[j]; }
  float nrm = fmaxf(sqrtf(q), 1e-12f);
#pragma unroll
  for (int j = 0; j < 5; ++j) U[r * 5 + j] = v[j] / nrm;
}

// z_out[r,c] = u[r, section(c)] * coms[r,c], sections: 500|500|2000|10
__global__ void k_zout(const float* __restrict__ coms, const float* __restrict__ U,
                       float* __restrict__ out, int rows)
{
  size_t i = (size_t)blockIdx.x * blockDim.x + threadIdx.x;
  size_t n = (size_t)rows * 3010;
  if (i >= n) return;
  int r = (int)(i / 3010), c = (int)(i % 3010);
  int sec = (c < 500) ? 0 : (c < 1000) ? 1 : (c < 3000) ? 2 : 3;
  out[i] = U[r * 5 + sec] * coms[i];
}

// ---------------------------------------------------------------------------
// Host-side launch helpers
// ---------------------------------------------------------------------------
static void gemm(const __bf16* A, const __bf16* BT, int Kpad,
                 const float* bias, float* C, int ldc,
                 int M, int Nc, int relu, hipStream_t s)
{
  dim3 g((Nc + 63) / 64, (M + 63) / 64), b(256);
  k_gemm_bf16_wmma<<<g, b, 0, s>>>(A, Kpad, BT, Kpad, bias, C, ldc,
                                   M, Kpad, Nc, relu);
}

// activation f32 [rows,cols](lds) -> padded bf16 [rows,pad32(cols)]
static void conv2bf(const float* src, int lds, int cols, __bf16* dst, int rows,
                    hipStream_t s)
{
  int Kpad = pad32(cols);
  size_t n = (size_t)rows * Kpad;
  k_f2bf_pad<<<dim3((unsigned)((n + 255) / 256)), dim3(256), 0, s>>>(
      src, lds, cols, dst, Kpad, rows);
}

static void spmm(const int* er, const int* ec, const float* ev,
                 const float* sup, float* agg, int nc, hipStream_t s)
{
  int bd = (nc >= 256) ? 256 : (((nc + 63) / 64) * 64);
  dim3 g(NEDGES, (nc + bd - 1) / bd), b(bd);
  k_spmm<<<g, b, 0, s>>>(er, ec, ev, sup, agg, nc);
}

static void act(float* x, size_t n, hipStream_t s)
{
  k_act<<<dim3((unsigned)((n + 255) / 256)), dim3(256), 0, s>>>(x, n);
}

// one GNN branch (4 layers). If coms != null, normalized layer outputs land in
// coms columns {0,500,1000,3000}; final layer also copied to out10.
static void run_branch(const __bf16* x_bf, __bf16* h_bf,
                       const __bf16* const Wbf[4],
                       const int* er, const int* ec, const float* ev,
                       float* bufA, float* bufB, float* coms, float* out10,
                       float* mean, float* rstd, hipStream_t s)
{
  static const int ind[4]  = {DIN, 500, 500, 2000};
  static const int outd[4] = {500, 500, 2000, 10};
  static const int cofs[4] = {0, 500, 1000, 3000};
  const __bf16* cur = x_bf;
  for (int l = 0; l < 4; ++l) {
    int K = ind[l], Nc = outd[l];
    gemm(cur, Wbf[l], pad32(K), nullptr, bufA, Nc, NNODES, Nc, 0, s);   // sup
    hipMemsetAsync(bufB, 0, (size_t)NNODES * Nc * sizeof(float), s);
    spmm(er, ec, ev, bufA, bufB, Nc, s);                                // agg
    act(bufB, (size_t)NNODES * Nc, s);                                  // elu∘lrelu
    k_colstats<<<dim3(Nc), dim3(256), 0, s>>>(bufB, Nc, NNODES, mean, rstd);
    float* dst; int ldd;
    if (coms)        { dst = coms + cofs[l]; ldd = 3010; }
    else if (l < 3)  { dst = bufB;           ldd = Nc;   }
    else             { dst = out10;          ldd = 10;   }
    size_t n = (size_t)NNODES * Nc;
    k_bnorm<<<dim3((unsigned)((n + 255) / 256)), dim3(256), 0, s>>>(
        bufB, Nc, dst, ldd, mean, rstd, NNODES, Nc);
    if (l < 3) {
      conv2bf(dst, ldd, Nc, h_bf, NNODES, s);
      cur = h_bf;
    } else if (coms) {
      size_t nn = (size_t)NNODES * 10;
      k_copy2d<<<dim3((unsigned)((nn + 255) / 256)), dim3(256), 0, s>>>(
          coms + 3000, 3010, out10, 10, NNODES, 10);
    }
  }
}

// ---------------------------------------------------------------------------
// kernel_launch
//   Workspace (~740 MB):
//     f32:  bufA[N*2000] bufB[N*2000] coms[N*3010] U[N*8] mean[3072] rstd[3072]
//     bf16: x_bf[N*1024] h_bf[N*3040] weight arena (~8M elems, transposed+padded)
// ---------------------------------------------------------------------------
extern "C" void kernel_launch(void* const* d_in, const int* in_sizes, int n_in,
                              void* d_out, int out_size, void* d_ws, size_t ws_size,
                              hipStream_t stream)
{
  (void)in_sizes; (void)n_in; (void)out_size; (void)ws_size;

  const float* x = (const float*)d_in[0];
  const int*   adj_r  = (const int*)d_in[1];
  const int*   adj_c  = (const int*)d_in[2];
  const float* adj_v  = (const float*)d_in[3];
  const int*   fadj_r = (const int*)d_in[4];
  const int*   fadj_c = (const int*)d_in[5];
  const float* fadj_v = (const float*)d_in[6];
  const int*   f1_r   = (const int*)d_in[7];
  const int*   f1_c   = (const int*)d_in[8];
  const float* f1_v   = (const float*)d_in[9];
  const int*   f2_r   = (const int*)d_in[10];
  const int*   f2_c   = (const int*)d_in[11];
  const float* f2_v   = (const float*)d_in[12];

  // ----- workspace carve-up -----
  float* ws = (float*)d_ws;
  size_t o = 0;
  float* bufA = ws + o; o += (size_t)NNODES * 2000;
  float* bufB = ws + o; o += (size_t)NNODES * 2000;
  float* coms = ws + o; o += (size_t)NNODES * 3010;
  float* Ubuf = ws + o; o += (size_t)NNODES * 8;
  float* mean = ws + o; o += 3072;
  float* rstd = ws + o; o += 3072;
  __bf16* bfp  = (__bf16*)(ws + o);
  __bf16* x_bf = bfp;                    bfp += (size_t)NNODES * 1024;  // pad32(1000)
  __bf16* h_bf = bfp;                    bfp += (size_t)NNODES * 3040;  // pad32(3010)
  __bf16* warena = bfp;

  // convert weight f32 [K,Nc] into transposed padded bf16 [pad64(Nc), pad32(K)]
  auto cw = [&](int idx, int K, int Nc) -> const __bf16* {
    __bf16* p = warena;
    int Kp = pad32(K), Np = pad64(Nc);
    size_t n = (size_t)Np * Kp;
    k_w2bfT<<<dim3((unsigned)((n + 255) / 256)), dim3(256), 0, stream>>>(
        (const float*)d_in[idx], K, Nc, p, Kp, Np);
    warena += n;
    return p;
  };

  static const int bdims[5] = {DIN, 500, 500, 2000, 10};
  const __bf16* feaW[4];
  const __bf16* topoW[4];
  for (int i = 0; i < 4; ++i) feaW[i]  = cw(13 + i, bdims[i], bdims[i + 1]);
  for (int i = 0; i < 4; ++i) topoW[i] = cw(17 + i, bdims[i], bdims[i + 1]);

  const __bf16* We1 = cw(21, DIN, 500);   const float* be1 = (const float*)d_in[22];
  const __bf16* We2 = cw(23, 500, 500);   const float* be2 = (const float*)d_in[24];
  const __bf16* We3 = cw(25, 500, 2000);  const float* be3 = (const float*)d_in[26];
  const __bf16* Wz1 = cw(27, 2000, 10);   const float* bz1 = (const float*)d_in[28];
  const __bf16* Wz2 = cw(29, 10, 10);     const float* bz2 = (const float*)d_in[30];
  const __bf16* Wd0 = cw(31, 10, 10);     const float* bd0 = (const float*)d_in[32];
  const __bf16* Wd1 = cw(33, 10, 2000);   const float* bd1 = (const float*)d_in[34];
  const __bf16* Wd2 = cw(35, 2000, 500);  const float* bd2 = (const float*)d_in[36];
  const __bf16* Wd3 = cw(37, 500, 500);   const float* bd3 = (const float*)d_in[38];
  const __bf16* Wxb = cw(39, 500, DIN);   const float* bxb = (const float*)d_in[40];
  const __bf16* Wml = cw(41, 3010, 5);    const float* bml = (const float*)d_in[42];

  // ----- output slices (return order) -----
  float* out = (float*)d_out;
  float* o_xbar   = out;
  float* o_z      = o_xbar   + (size_t)NNODES * 1000;
  float* o_fea3   = o_z      + (size_t)NNODES * 10;
  float* o_topo3  = o_fea3   + (size_t)NNODES * 10;
  float* o_fead3  = o_topo3  + (size_t)NNODES * 10;
  float* o_topod3 = o_fead3  + (size_t)NNODES * 10;
  float* o_zout   = o_topod3 + (size_t)NNODES * 10;

  // ----- x -> bf16 (shared by AE + all branches) -----
  conv2bf(x, DIN, DIN, x_bf, NNODES, stream);

  // ----- AE branch -----
  gemm(x_bf, We1, 1024, be1, bufA, 500, NNODES, 500, 1, stream);        // h1
  conv2bf(bufA, 500, 500, h_bf, NNODES, stream);
  gemm(h_bf, We2, 512, be2, bufB, 500, NNODES, 500, 1, stream);         // h2
  conv2bf(bufB, 500, 500, h_bf, NNODES, stream);
  gemm(h_bf, We3, 512, be3, bufA, 2000, NNODES, 2000, 1, stream);       // h3
  conv2bf(bufA, 2000, 2000, h_bf, NNODES, stream);
  gemm(h_bf, Wz1, 2048, bz1, bufB, 10, NNODES, 10, 0, stream);          // z1
  conv2bf(bufB, 10, 10, h_bf, NNODES, stream);
  gemm(h_bf, Wz2, 32, bz2, o_z, 10, NNODES, 10, 0, stream);             // z -> out
  conv2bf(o_z, 10, 10, h_bf, NNODES, stream);
  gemm(h_bf, Wd0, 32, bd0, bufA, 10, NNODES, 10, 1, stream);            // d
  conv2bf(bufA, 10, 10, h_bf, NNODES, stream);
  gemm(h_bf, Wd1, 32, bd1, bufB, 2000, NNODES, 2000, 1, stream);
  conv2bf(bufB, 2000, 2000, h_bf, NNODES, stream);
  gemm(h_bf, Wd2, 2048, bd2, bufA, 500, NNODES, 500, 1, stream);
  conv2bf(bufA, 500, 500, h_bf, NNODES, stream);
  gemm(h_bf, Wd3, 512, bd3, bufB, 500, NNODES, 500, 1, stream);
  conv2bf(bufB, 500, 500, h_bf, NNODES, stream);
  gemm(h_bf, Wxb, 512, bxb, o_xbar, DIN, NNODES, DIN, 0, stream);       // x_bar

  // ----- GNN branches -----
  run_branch(x_bf, h_bf, feaW,  fadj_r, fadj_c, fadj_v, bufA, bufB, coms,    o_fea3,   mean, rstd, stream);
  run_branch(x_bf, h_bf, topoW, adj_r,  adj_c,  adj_v,  bufA, bufB, nullptr, o_topo3,  mean, rstd, stream);
  run_branch(x_bf, h_bf, feaW,  f2_r,   f2_c,   f2_v,   bufA, bufB, nullptr, o_fead3,  mean, rstd, stream);
  run_branch(x_bf, h_bf, topoW, f1_r,   f1_c,   f1_v,   bufA, bufB, nullptr, o_topod3, mean, rstd, stream);

  // ----- attention head + z_out_fea -----
  conv2bf(coms, 3010, 3010, h_bf, NNODES, stream);
  gemm(h_bf, Wml, 3040, bml, Ubuf, 5, NNODES, 5, 0, stream);
  k_head<<<dim3((NNODES + 255) / 256), dim3(256), 0, stream>>>(Ubuf, NNODES);
  {
    size_t n = (size_t)NNODES * 3010;
    k_zout<<<dim3((unsigned)((n + 255) / 256)), dim3(256), 0, stream>>>(
        coms, Ubuf, o_zout, NNODES);
  }
}
